// FashionResnet_54451595379187
// MI455X (gfx1250) — matrix operand order repaired and checked by reading, not compiled, not run on hardware
//
#include <hip/hip_runtime.h>
#include <hip/hip_bf16.h>
#include <math.h>

// ---------------------------------------------------------------------------
// CDNA5 (gfx1250) WMMA types
// ---------------------------------------------------------------------------
typedef __attribute__((ext_vector_type(16))) _Float16 v16h;
typedef __attribute__((ext_vector_type(8)))  float    v8f;

union FragA { v16h v; _Float16 h[16]; };
union FragC { v8f  v; float    f[8];  };

#define TB 64          // batch
#define SPD 7
#define ROIW 3
#define OUTD 512

// Async global->LDS path (gfx1250). Device-pass only; guarded so a toolchain
// without the builtin falls back to scalar LDS fills and still compiles.
// Probe-confirmed signature: (v4i __device__*, v4i __shared__*, imm int, imm int)
#if defined(__HIP_DEVICE_COMPILE__) && defined(__gfx1250__) && \
    __has_builtin(__builtin_amdgcn_global_load_async_to_lds_b128)
#define ASYNC_LDS 1
typedef int v4i_ __attribute__((ext_vector_type(4)));
typedef __attribute__((address_space(1))) v4i_* gv4p;   // global (AS1)
typedef __attribute__((address_space(3))) v4i_* sv4p;   // LDS (AS3)
#else
#define ASYNC_LDS 0
#endif

__device__ __forceinline__ void async_wait0()
{
#if ASYNC_LDS
#if __has_builtin(__builtin_amdgcn_s_wait_asynccnt)
    __builtin_amdgcn_s_wait_asynccnt(0);
#else
    asm volatile("s_wait_asynccnt 0x0" ::: "memory");
#endif
#endif
}

// ---------------------------------------------------------------------------
// Implicit-GEMM convolution on v_wmma_f32_16x16x32_f16.
//   activations: NHWC f16, weights: [K][Cout] f16, K = (kh*KW+kw)*Cin+cin
//   GEMM: M = B*OH*OW, N = Cout, K = Cin*KH*KW
//   128 threads = 4 waves; block tile 64x64; wave tile 32x32 (4 wmma/K-step).
//   Epilogue fuses folded BN (g,b,m,v) or plain bias, residual add, ReLU.
// ---------------------------------------------------------------------------
__global__ __launch_bounds__(128)
void k_conv_wmma(const _Float16* __restrict__ act,
                 const _Float16* __restrict__ wt,
                 _Float16* __restrict__ outh,        // may be null
                 float*    __restrict__ outf,        // may be null
                 const _Float16* __restrict__ resid, // may be null (NHWC, Cout ch)
                 const float* __restrict__ bng, const float* __restrict__ bnb,
                 const float* __restrict__ bnm, const float* __restrict__ bnv,
                 const float* __restrict__ bias,     // may be null
                 int Bn, int H, int W, int Cin, int Cout,
                 int KH, int KW, int stride, int pad,
                 int OH, int OW, int relu)
{
    // Row strides 40 and 72 halves => 80/144 bytes: multiples of 16B, so every
    // 8-half chunk written by async b128 DMA is 16-byte aligned.
    __shared__ __align__(16) _Float16 As[64][40];   // 64 M x 32 K
    __shared__ __align__(16) _Float16 Bs[32][72];   // 32 K x 64 N

    const int tid   = threadIdx.x;
    const int lane  = tid & 31;
    const int wave  = tid >> 5;
    const int waveM = wave & 1;          // 2 x 2 wave grid
    const int waveN = wave >> 1;
    const int half  = lane >> 4;

    const int Mtot = Bn * OH * OW;
    const int Ktot = Cin * KH * KW;
    const int m0   = blockIdx.x * 64;
    const int n0   = blockIdx.y * 64;

    // fast (dense, aligned) tile conditions
    const bool oneByOne = (KH == 1) && (KW == 1) && (stride == 1) && (pad == 0);
    const bool fastA = oneByOne && ((Cin & 7) == 0) && (m0 + 64 <= Mtot);
    const bool fastB = ((Cout & 7) == 0) && (n0 + 64 <= Cout);

    FragC acc[2][2];
    #pragma unroll
    for (int mi = 0; mi < 2; ++mi)
        #pragma unroll
        for (int ni = 0; ni < 2; ++ni)
            acc[mi][ni].v = (v8f){0.f,0.f,0.f,0.f,0.f,0.f,0.f,0.f};

    for (int k0 = 0; k0 < Ktot; k0 += 32) {
        const bool ktFull = (k0 + 32 <= Ktot);
        bool usedAsync = false;

        // ---------------- A tile: 64 M x 32 K ----------------
#if ASYNC_LDS
        if (fastA && ktFull) {
            // 64 rows x 4 chunks of 8 halves = 256 b128 DMAs, 2 per thread
            #pragma unroll
            for (int it = 0; it < 2; ++it) {
                int idx = tid + it * 128;
                int m = idx >> 2, c = idx & 3;
                const _Float16* src = act + (size_t)(m0 + m) * Cin + k0 + c * 8;
                __builtin_amdgcn_global_load_async_to_lds_b128(
                    (gv4p)src, (sv4p)&As[m][c * 8], 0, 0);
            }
            usedAsync = true;
        } else
#endif
        {
            #pragma unroll
            for (int it = 0; it < 16; ++it) {
                int t  = tid + it * 128;          // 0..2047
                int m  = t >> 5;
                int k  = t & 31;
                int gm = m0 + m;
                int gk = k0 + k;
                _Float16 v = (_Float16)0.0f;
                if (gm < Mtot && gk < Ktot) {
                    int cin = gk % Cin;
                    int rk  = gk / Cin;
                    int kw  = rk % KW;
                    int kh  = rk / KW;
                    int pix = gm % (OH * OW);
                    int b   = gm / (OH * OW);
                    int oh  = pix / OW;
                    int ow  = pix % OW;
                    int ih  = oh * stride - pad + kh;
                    int iw  = ow * stride - pad + kw;
                    if (ih >= 0 && ih < H && iw >= 0 && iw < W)
                        v = act[(((size_t)b * H + ih) * W + iw) * Cin + cin];
                }
                As[m][k] = v;
            }
        }

        // ---------------- B tile: 32 K x 64 N ----------------
#if ASYNC_LDS
        if (fastB && ktFull) {
            // 32 rows x 8 chunks of 8 halves = 256 b128 DMAs, 2 per thread
            #pragma unroll
            for (int it = 0; it < 2; ++it) {
                int idx = tid + it * 128;
                int k = idx >> 3, c = idx & 7;
                const _Float16* src = wt + (size_t)(k0 + k) * Cout + n0 + c * 8;
                __builtin_amdgcn_global_load_async_to_lds_b128(
                    (gv4p)src, (sv4p)&Bs[k][c * 8], 0, 0);
            }
            usedAsync = true;
        } else
#endif
        {
            #pragma unroll
            for (int it = 0; it < 16; ++it) {
                int t  = tid + it * 128;
                int k  = t >> 6;
                int n  = t & 63;
                int gk = k0 + k;
                int gn = n0 + n;
                _Float16 v = (_Float16)0.0f;
                if (gk < Ktot && gn < Cout)
                    v = wt[(size_t)gk * Cout + gn];
                Bs[k][n] = v;
            }
        }

        // prefetch next weight K-slice into cache (global_prefetch_b8)
        if (k0 + 32 < Ktot)
            __builtin_prefetch(wt + (size_t)(k0 + 32) * Cout + n0, 0, 1);

        if (usedAsync) async_wait0();
        __syncthreads();

        // ---------------- fragments (CDNA5 wave32 layouts) ----------------
        FragA a[2], b[2];
        #pragma unroll
        for (int mi = 0; mi < 2; ++mi) {
            const int mrow = waveM * 32 + mi * 16 + (lane & 15);
            #pragma unroll
            for (int e = 0; e < 8; ++e)  a[mi].h[e] = As[mrow][8 * half + e];
            #pragma unroll
            for (int e = 8; e < 16; ++e) a[mi].h[e] = As[mrow][16 + 8 * half + (e - 8)];
        }
        #pragma unroll
        for (int ni = 0; ni < 2; ++ni) {
            const int ncol = waveN * 32 + ni * 16 + (lane & 15);
            #pragma unroll
            for (int e = 0; e < 8; ++e)  b[ni].h[e] = Bs[8 * half + e][ncol];
            #pragma unroll
            for (int e = 8; e < 16; ++e) b[ni].h[e] = Bs[16 + 8 * half + (e - 8)][ncol];
        }

        #pragma unroll
        for (int mi = 0; mi < 2; ++mi)
            #pragma unroll
            for (int ni = 0; ni < 2; ++ni)
                acc[mi][ni].v = __builtin_amdgcn_wmma_f32_16x16x32_f16(
                    false, a[mi].v, false, b[ni].v, (short)0, acc[mi][ni].v,
                    false, false);
        __syncthreads();
    }

    // ---------------- epilogue: D layout M = r + 8*half, N = lane&15 ----------
    #pragma unroll
    for (int ni = 0; ni < 2; ++ni) {
        const int n = n0 + waveN * 32 + ni * 16 + (lane & 15);
        if (n >= Cout) continue;

        float scale = 1.0f, shift = 0.0f;
        if (bng) {
            scale = bng[n] * rsqrtf(bnv[n] + 1e-5f);
            shift = bnb[n] - bnm[n] * scale;
        } else if (bias) {
            shift = bias[n];
        }

        #pragma unroll
        for (int mi = 0; mi < 2; ++mi) {
            #pragma unroll
            for (int r = 0; r < 8; ++r) {
                int gm = m0 + waveM * 32 + mi * 16 + half * 8 + r;
                if (gm >= Mtot) continue;
                float v = acc[mi][ni].f[r] * scale + shift;
                if (resid) v += (float)resid[(size_t)gm * Cout + n];
                if (relu)  v = fmaxf(v, 0.0f);
                size_t oi = (size_t)gm * Cout + n;
                if (outh) outh[oi] = (_Float16)v;
                if (outf) outf[oi] = v;
            }
        }
    }
}

// ---------------------------------------------------------------------------
// Small helper kernels
// ---------------------------------------------------------------------------
__global__ void k_f32_to_f16(const float* __restrict__ in, _Float16* __restrict__ out, int n)
{
    int i = blockIdx.x * blockDim.x + threadIdx.x;
    if (i < n) out[i] = (_Float16)in[i];
}

// x (B,3,224,224) f32 NCHW -> NHWC f16
__global__ void k_nchw_to_nhwc(const float* __restrict__ x, _Float16* __restrict__ out,
                               int Bn, int C, int H, int W)
{
    int i = blockIdx.x * blockDim.x + threadIdx.x;
    int tot = Bn * C * H * W;
    if (i >= tot) return;
    int c = i % C; int t = i / C;
    int w = t % W; t /= W;
    int h = t % H; int b = t / H;
    out[i] = (_Float16)x[(((size_t)b * C + c) * H + h) * W + w];
}

// OIHW f32 -> [K][Cout] f16,  K = (kh*KW+kw)*Cin + cin
__global__ void k_repack_oihw(const float* __restrict__ w, _Float16* __restrict__ out,
                              int Cout, int Cin, int KH, int KW)
{
    int i = blockIdx.x * blockDim.x + threadIdx.x;
    int tot = Cout * Cin * KH * KW;
    if (i >= tot) return;
    int o  = i / (Cin * KH * KW);
    int r  = i % (Cin * KH * KW);
    int ci = r / (KH * KW);
    int r2 = r % (KH * KW);
    int kh = r2 / KW, kw = r2 % KW;
    out[((size_t)((kh * KW + kw) * Cin + ci)) * Cout + o] = (_Float16)w[i];
}

// NHWC f16 maxpool 3x3 s2 pad1: (B,112,112,64) -> (B,56,56,64)
__global__ void k_maxpool(const _Float16* __restrict__ in, _Float16* __restrict__ out,
                          int Bn, int H, int W, int C, int OH, int OW)
{
    int i = blockIdx.x * blockDim.x + threadIdx.x;
    int tot = Bn * OH * OW * C;
    if (i >= tot) return;
    int c = i % C; int t = i / C;
    int ow = t % OW; t /= OW;
    int oh = t % OH; int b = t / OH;
    float m = -3.0e38f;
    #pragma unroll
    for (int kh = 0; kh < 3; ++kh) {
        int ih = oh * 2 - 1 + kh;
        if (ih < 0 || ih >= H) continue;
        #pragma unroll
        for (int kw = 0; kw < 3; ++kw) {
            int iw = ow * 2 - 1 + kw;
            if (iw < 0 || iw >= W) continue;
            float v = (float)in[(((size_t)b * H + ih) * W + iw) * C + c];
            m = fmaxf(m, v);
        }
    }
    out[i] = (_Float16)m;
}

// fc_bbox: pre (B,7,7,32 NHWC f16) flattened as NCHW c*49+s  -> sigmoid -> bbox, rois
__global__ __launch_bounds__(128)
void k_fc_bbox(const _Float16* __restrict__ pre, const float* __restrict__ w,
               const float* __restrict__ bias, float* __restrict__ bbox,
               int* __restrict__ rois)
{
    __shared__ float red[128 * 4];
    __shared__ float sig[4];
    int b = blockIdx.x;
    float a0 = 0.f, a1 = 0.f, a2 = 0.f, a3 = 0.f;
    for (int d = threadIdx.x; d < 32 * SPD * SPD; d += 128) {
        int c = d / 49; int s = d - c * 49;
        int oh = s / 7, ow = s - (s / 7) * 7;
        float xv = (float)pre[(((size_t)b * SPD + oh) * SPD + ow) * 32 + c];
        const float* wr = w + (size_t)d * 4;
        a0 += xv * wr[0]; a1 += xv * wr[1]; a2 += xv * wr[2]; a3 += xv * wr[3];
    }
    red[threadIdx.x * 4 + 0] = a0; red[threadIdx.x * 4 + 1] = a1;
    red[threadIdx.x * 4 + 2] = a2; red[threadIdx.x * 4 + 3] = a3;
    __syncthreads();
    if (threadIdx.x < 4) {
        float s = bias[threadIdx.x];
        for (int t = 0; t < 128; ++t) s += red[t * 4 + threadIdx.x];
        sig[threadIdx.x] = 1.0f / (1.0f + expf(-s));
    }
    __syncthreads();
    if (threadIdx.x == 0) {
        float b0 = sig[0], b1 = sig[1];
        float b2 = b0 + sig[2], b3 = b1 + sig[3];
        bbox[b * 4 + 0] = b0; bbox[b * 4 + 1] = b1;
        bbox[b * 4 + 2] = b2; bbox[b * 4 + 3] = b3;
        rois[b * 4 + 0] = (int)floorf(b0 * (float)SPD);
        rois[b * 4 + 1] = (int)floorf(b1 * (float)SPD);
        rois[b * 4 + 2] = (int)floorf(b2 * (float)SPD);
        rois[b * 4 + 3] = (int)floorf(b3 * (float)SPD);
    }
}

// global mean of feat (B,7,7,512 NHWC) -> allf[b*5120 + c]
__global__ void k_global_mean(const _Float16* __restrict__ feat, _Float16* __restrict__ allf)
{
    int i = blockIdx.x * blockDim.x + threadIdx.x;
    if (i >= TB * OUTD) return;
    int b = i / OUTD, c = i % OUTD;
    float s = 0.f;
    for (int p = 0; p < SPD * SPD; ++p)
        s += (float)feat[((size_t)b * SPD * SPD + p) * OUTD + c];
    allf[(size_t)b * (OUTD + ROIW * ROIW * OUTD) + c] = (_Float16)(s / (float)(SPD * SPD));
}

// ROI adaptive max 3x3 -> allf[b*5120 + 512 + c*9 + ph*3 + pw]
__global__ void k_roi_pool(const _Float16* __restrict__ feat, const int* __restrict__ rois,
                           _Float16* __restrict__ allf)
{
    int i = blockIdx.x * blockDim.x + threadIdx.x;
    if (i >= TB * OUTD) return;
    int b = i / OUTD, c = i % OUTD;
    int x1 = rois[b * 4 + 0], y1 = rois[b * 4 + 1];
    int x2 = rois[b * 4 + 2]; if (x2 > SPD - 1) x2 = SPD - 1;
    int y2 = rois[b * 4 + 3]; if (y2 > SPD - 1) y2 = SPD - 1;
    int Hh = y2 - y1 + 1, Ww = x2 - x1 + 1;
    size_t base = (size_t)b * (OUTD + ROIW * ROIW * OUTD) + OUTD + (size_t)c * 9;
    for (int ph = 0; ph < ROIW; ++ph) {
        int hs = (ph * Hh) / ROIW;
        int he = ((ph + 1) * Hh + ROIW - 1) / ROIW;
        int ya = y1 + hs; if (ya < 0) ya = 0;
        int yb = y1 + he - 1; if (yb > SPD - 1) yb = SPD - 1;
        for (int pw = 0; pw < ROIW; ++pw) {
            int ws = (pw * Ww) / ROIW;
            int we = ((pw + 1) * Ww + ROIW - 1) / ROIW;
            int xa = x1 + ws; if (xa < 0) xa = 0;
            int xb = x1 + we - 1; if (xb > SPD - 1) xb = SPD - 1;
            float m = -3.0e38f;
            for (int y = ya; y <= yb; ++y)
                for (int x = xa; x <= xb; ++x)
                    m = fmaxf(m, (float)feat[(((size_t)b * SPD + y) * SPD + x) * OUTD + c]);
            allf[base + ph * 3 + pw] = (_Float16)m;
        }
    }
}

// ---------------------------------------------------------------------------
// Host-side launch
// ---------------------------------------------------------------------------
extern "C" void kernel_launch(void* const* d_in, const int* in_sizes, int n_in,
                              void* d_out, int out_size, void* d_ws, size_t ws_size,
                              hipStream_t stream)
{
    (void)in_sizes; (void)n_in; (void)out_size; (void)ws_size;
    auto F = [&](int i) { return (const float*)d_in[i]; };

    // JAX pytree order (sorted dict keys; lists in order):
    // 0 fc_bbox.b  1 fc_bbox.w  2 fc_bin.b  3 fc_bin.w  4 fc_cls.b  5 fc_cls.w
    // layers at 6..100 (block: bn1{b,g,m,v}, bn2{b,g,m,v}, [dbn{b,g,m,v}, dw], w1, w2)
    // 101 reduce_b  102 reduce_w  103..106 stem_bn{b,g,m,v}  107 stem_w  108 x
    const float* fcbb_b = F(0);  const float* fcbb_w = F(1);
    const float* fbin_b = F(2);  const float* fbin_w = F(3);
    const float* fcls_b = F(4);  const float* fcls_w = F(5);
    const float* reduce_b = F(101); const float* reduce_w = F(102);
    const float* stem_g = F(104), *stem_b = F(103), *stem_m = F(105), *stem_v = F(106);
    const float* stem_w = F(107);
    const float* x = F(108);

    struct Blk { int base, cin, cout, stride, hasdw; };
    Blk blk[8] = {
        {  6,  64,  64, 1, 0 }, { 16,  64,  64, 1, 0 },
        { 26,  64, 128, 2, 1 }, { 41, 128, 128, 1, 0 },
        { 51, 128, 256, 2, 1 }, { 66, 256, 256, 1, 0 },
        { 76, 256, 512, 2, 1 }, { 91, 512, 512, 1, 0 },
    };

    // ---- workspace bump allocator ----
    size_t off = 0;
    auto alloc = [&](size_t bytes) {
        void* p = (char*)d_ws + off;
        off = (off + bytes + 255) & ~(size_t)255;
        return p;
    };
    _Float16* xh   = (_Float16*)alloc((size_t)TB * 224 * 224 * 3 * 2);
    _Float16* sOut = (_Float16*)alloc((size_t)TB * 112 * 112 * 64 * 2);
    size_t bufE = (size_t)TB * 56 * 56 * 64;       // max per-layer activation elems
    _Float16* buf0 = (_Float16*)alloc(bufE * 2);
    _Float16* buf1 = (_Float16*)alloc(bufE * 2);
    _Float16* buf2 = (_Float16*)alloc(bufE * 2);
    _Float16* redOut = (_Float16*)alloc((size_t)TB * SPD * SPD * 32 * 2);
    size_t allfD = OUTD + ROIW * ROIW * OUTD;      // 5120
    _Float16* allf = (_Float16*)alloc((size_t)TB * allfD * 2);
    int* rois = (int*)alloc(TB * 4 * sizeof(int));

    auto allocW = [&](size_t elems) { return (_Float16*)alloc(elems * 2); };

    // ---- kernel helpers ----
    auto repack = [&](const float* w, _Float16* ww, int Cout, int Cin, int KH, int KW) {
        int n = Cout * Cin * KH * KW;
        k_repack_oihw<<<(n + 255) / 256, 256, 0, stream>>>(w, ww, Cout, Cin, KH, KW);
    };
    auto toh = [&](const float* w, _Float16* ww, int n) {
        k_f32_to_f16<<<(n + 255) / 256, 256, 0, stream>>>(w, ww, n);
    };
    auto conv = [&](const _Float16* act, const _Float16* wt,
                    _Float16* outh, float* outf, const _Float16* resid,
                    const float* g, const float* bb, const float* mm, const float* vv,
                    const float* bias,
                    int Bn, int H, int W, int Cin, int Cout,
                    int KH, int KW, int s, int p, int relu) {
        int OH = (H + 2 * p - KH) / s + 1;
        int OW = (W + 2 * p - KW) / s + 1;
        int M = Bn * OH * OW;
        dim3 grid((M + 63) / 64, (Cout + 63) / 64);
        k_conv_wmma<<<grid, 128, 0, stream>>>(act, wt, outh, outf, resid,
                                              g, bb, mm, vv, bias,
                                              Bn, H, W, Cin, Cout, KH, KW, s, p, OH, OW, relu);
    };

    // ---- repack / convert all weights (f32 -> f16) ----
    _Float16* wStem = allocW(64 * 3 * 49);     repack(stem_w, wStem, 64, 3, 7, 7);
    _Float16* wB[8][3];                         // [block]{w1, w2, dw}
    for (int i = 0; i < 8; ++i) {
        const Blk& B = blk[i];
        int wbase = B.base + (B.hasdw ? 12 : 8); // after bn1(4), bn2(4), [dbn(4)]
        const float* w1 = B.hasdw ? F(wbase + 2) : F(wbase + 0);
        const float* w2 = B.hasdw ? F(wbase + 3) : F(wbase + 1);
        wB[i][0] = allocW((size_t)B.cout * B.cin * 9);  repack(w1, wB[i][0], B.cout, B.cin, 3, 3);
        wB[i][1] = allocW((size_t)B.cout * B.cout * 9); repack(w2, wB[i][1], B.cout, B.cout, 3, 3);
        wB[i][2] = nullptr;
        if (B.hasdw) {
            const float* dw = F(wbase + 1);
            wB[i][2] = allocW((size_t)B.cout * B.cin);  repack(dw, wB[i][2], B.cout, B.cin, 1, 1);
        }
    }
    _Float16* wRed = allocW(512 * 32);            repack(reduce_w, wRed, 32, 512, 1, 1);
    _Float16* wBin = allocW((size_t)allfD * 1000); toh(fbin_w, wBin, (int)(allfD * 1000));
    _Float16* wCls = allocW((size_t)allfD * 50);   toh(fcls_w, wCls, (int)(allfD * 50));

    // ---- input NCHW f32 -> NHWC f16 ----
    {
        int n = TB * 3 * 224 * 224;
        k_nchw_to_nhwc<<<(n + 255) / 256, 256, 0, stream>>>(x, xh, TB, 3, 224, 224);
    }

    // ---- stem: conv7x7 s2 + BN + ReLU, then maxpool 3x3 s2 ----
    conv(xh, wStem, sOut, nullptr, nullptr, stem_g, stem_b, stem_m, stem_v, nullptr,
         TB, 224, 224, 3, 64, 7, 7, 2, 3, 1);
    {
        int n = TB * 56 * 56 * 64;
        k_maxpool<<<(n + 255) / 256, 256, 0, stream>>>(sOut, buf0, TB, 112, 112, 64, 56, 56);
    }

    // ---- residual blocks ----
    _Float16* cur = buf0;           // current activation
    int H = 56, W = 56;
    int sizes[4] = { 56, 28, 14, 7 };
    for (int i = 0; i < 8; ++i) {
        const Blk& B = blk[i];
        int li = i / 2;
        int OHW = sizes[li];
        const float* g1 = F(B.base + 1), *b1 = F(B.base + 0), *m1 = F(B.base + 2), *v1 = F(B.base + 3);
        const float* g2 = F(B.base + 5), *b2 = F(B.base + 4), *m2 = F(B.base + 6), *v2 = F(B.base + 7);
        _Float16* y  = (cur == buf1) ? buf2 : buf1;            // conv1 output
        _Float16* sc = cur;                                    // skip source
        _Float16* dst;
        if (B.hasdw) {
            const float* gd = F(B.base + 9), *bd = F(B.base + 8), *md = F(B.base + 10), *vd = F(B.base + 11);
            _Float16* dsb = (cur != buf0 && y != buf0) ? buf0 : ((cur != buf2 && y != buf2) ? buf2 : buf1);
            conv(cur, wB[i][2], dsb, nullptr, nullptr, gd, bd, md, vd, nullptr,
                 TB, H, W, B.cin, B.cout, 1, 1, B.stride, 0, 0);
            sc = dsb;
        }
        conv(cur, wB[i][0], y, nullptr, nullptr, g1, b1, m1, v1, nullptr,
             TB, H, W, B.cin, B.cout, 3, 3, B.stride, 1, 1);
        // conv2 + BN + residual + ReLU; write to a buffer != y and != sc
        dst = (y != buf0 && sc != buf0) ? buf0 : ((y != buf1 && sc != buf1) ? buf1 : buf2);
        conv(y, wB[i][1], dst, nullptr, sc, g2, b2, m2, v2, nullptr,
             TB, OHW, OHW, B.cout, B.cout, 3, 3, 1, 1, 1);
        cur = dst;
        H = OHW; W = OHW;
    }
    _Float16* feat = cur;           // (B, 7, 7, 512) NHWC f16

    // ---- head ----
    conv(feat, wRed, redOut, nullptr, nullptr, nullptr, nullptr, nullptr, nullptr, reduce_b,
         TB, SPD, SPD, OUTD, 32, 1, 1, 1, 0, 0);

    float* out_cls  = (float*)d_out;                       // (64, 50)
    float* out_bin  = (float*)d_out + TB * 50;             // (64, 1000)
    float* out_bbox = (float*)d_out + TB * 50 + TB * 1000; // (64, 4)

    k_fc_bbox<<<TB, 128, 0, stream>>>(redOut, fcbb_w, fcbb_b, out_bbox, rois);

    {
        int n = TB * OUTD;
        k_global_mean<<<(n + 255) / 256, 256, 0, stream>>>(feat, allf);
        k_roi_pool<<<(n + 255) / 256, 256, 0, stream>>>(feat, rois, allf);
    }

    // final FCs as 1x1 "convs" over allf (M=64, K=5120): WMMA GEMMs, f32 out
    conv(allf, wCls, nullptr, out_cls, nullptr, nullptr, nullptr, nullptr, nullptr, fcls_b,
         TB, 1, 1, (int)allfD, 50, 1, 1, 1, 0, 0);
    conv(allf, wBin, nullptr, out_bin, nullptr, nullptr, nullptr, nullptr, nullptr, fbin_b,
         TB, 1, 1, (int)allfD, 1000, 1, 1, 1, 0, 0);
}
